// MultiBandSparseModel_51599737094442
// MI455X (gfx1250) — matrix operand order repaired and checked by reading, not compiled, not run on hardware
//
#include <hip/hip_runtime.h>
#include <cmath>

// ---------------------------------------------------------------------------
// MultiBandSparseModel for MI455X (gfx1250).
// Heavy op: dictionary conv as f16 WMMA GEMM (v_wmma_f32_16x16x32_f16),
// register-resident A fragments, 8 column tiles per wave, LDS-staged signal
// window (async-to-LDS when available), transposed b128 C writeback.
// FFT machinery: global-memory radix-2 stages (in-place, bit-reversed DIT).
// ---------------------------------------------------------------------------

typedef __attribute__((ext_vector_type(16))) _Float16 v16h;
typedef __attribute__((ext_vector_type(8)))  float    v8f;
typedef __attribute__((ext_vector_type(4)))  int      v4i;

struct cplx { float x, y; };

#define N_SAMP 32768
#define N_ATOMS 512
#define KERN 256
#define K_KEEP 64
#define BATCH 4
#define HID 128
#define FEAT_IN 3072
#define NT 8                      // column tiles per wave
#define SXW (16 * NT + KERN)      // LDS window halves = 384

#if defined(__gfx1250__) && __has_builtin(__builtin_amdgcn_global_load_async_to_lds_b128) && __has_builtin(__builtin_amdgcn_s_wait_asynccnt)
#define USE_ASYNC_LDS 1
#else
#define USE_ASYNC_LDS 0
#endif

// ------------------------------- FFT kernels -------------------------------

__global__ __launch_bounds__(256) void k_r2c(const float* __restrict__ src,
                                             cplx* __restrict__ dst, int n, int m) {
  int i = blockIdx.x * 256 + threadIdx.x;
  if (i >= m) return;
  cplx v; v.x = (i < n) ? src[i] : 0.f; v.y = 0.f;
  dst[i] = v;
}

__global__ __launch_bounds__(256) void k_bitrev(cplx* __restrict__ a, int n, int lg) {
  int i = blockIdx.x * 256 + threadIdx.x;
  if (i >= n) return;
  int j = (int)(__brev((unsigned)i) >> (32 - lg));
  if (i < j) { cplx t = a[i]; a[i] = a[j]; a[j] = t; }
}

__global__ __launch_bounds__(256) void k_fft_stage(cplx* __restrict__ a, int n,
                                                   int len, float sign) {
  int t = blockIdx.x * 256 + threadIdx.x;
  if (t >= (n >> 1)) return;
  int half = len >> 1;
  int blk = t / half, j = t - blk * half;
  int i0 = blk * len + j, i1 = i0 + half;
  float ang = sign * 6.28318530717958647692f * (float)j / (float)len;
  float sn, cs; __sincosf(ang, &sn, &cs);
  cplx u = a[i0], v = a[i1];
  cplx w; w.x = v.x * cs - v.y * sn; w.y = v.x * sn + v.y * cs;
  cplx o0; o0.x = u.x + w.x; o0.y = u.y + w.y;
  cplx o1; o1.x = u.x - w.x; o1.y = u.y - w.y;
  a[i0] = o0; a[i1] = o1;
}

__global__ __launch_bounds__(256) void k_band_spec(const cplx* __restrict__ X,
                                                   cplx* __restrict__ out, int s,
                                                   int mn, int mx, float scale) {
  int i = blockIdx.x * 256 + threadIdx.x;
  if (i >= s) return;
  int half = s >> 1;
  int k = (i <= half) ? i : (s - i);
  cplx v; v.x = 0.f; v.y = 0.f;
  if (k >= mn && k < mx) { v.x = X[k].x * scale; v.y = X[k].y * scale; }
  if (i > half) v.y = -v.y;
  out[i] = v;
}

__global__ __launch_bounds__(256) void k_c2r_scale(float* __restrict__ dst,
                                                   const cplx* __restrict__ src,
                                                   int n, float scale) {
  int i = blockIdx.x * 256 + threadIdx.x;
  if (i < n) dst[i] = src[i].x * scale;
}

__global__ __launch_bounds__(256) void k_zero_c(cplx* __restrict__ a, int n) {
  int i = blockIdx.x * 256 + threadIdx.x;
  if (i < n) { cplx z; z.x = 0.f; z.y = 0.f; a[i] = z; }
}

__global__ __launch_bounds__(256) void k_spec_accum(const cplx* __restrict__ S,
                                                    cplx* __restrict__ acc, int s,
                                                    float scale) {
  int k = blockIdx.x * 256 + threadIdx.x;
  if (k > (s >> 1)) return;
  cplx a = acc[k];
  a.x += S[k].x * scale; a.y += S[k].y * scale;
  acc[k] = a;
}

__global__ __launch_bounds__(256) void k_herm(const cplx* __restrict__ accHalf,
                                              cplx* __restrict__ buf, int n) {
  int i = blockIdx.x * 256 + threadIdx.x;
  if (i >= n) return;
  int half = n >> 1;
  if (i <= half) buf[i] = accHalf[i];
  else { cplx v = accHalf[n - i]; v.y = -v.y; buf[i] = v; }
}

__global__ __launch_bounds__(256) void k_signal(const cplx* __restrict__ buf,
                                                const float* __restrict__ topBand,
                                                float* __restrict__ sig, int n,
                                                float scale) {
  int i = blockIdx.x * 256 + threadIdx.x;
  if (i < n) sig[i] = buf[i].x * scale + topBand[i];
}

__global__ __launch_bounds__(256) void k_cmul(cplx* __restrict__ a,
                                              const cplx* __restrict__ b, int n,
                                              float scale) {
  int i = blockIdx.x * 256 + threadIdx.x;
  if (i >= n) return;
  cplx u = a[i], v = b[i];
  cplx o; o.x = (u.x * v.x - u.y * v.y) * scale; o.y = (u.x * v.y + u.y * v.x) * scale;
  a[i] = o;
}

__global__ __launch_bounds__(256) void k_out(const float* __restrict__ sig,
                                             const cplx* __restrict__ wet,
                                             const float* __restrict__ mix,
                                             float* __restrict__ out, int n,
                                             float scale) {
  int i = blockIdx.x * 256 + threadIdx.x;
  if (i >= n) return;
  float m = mix[0];
  out[i] = sig[i] * m + (wet[i].x * scale) * (1.f - m);
}

// ------------------------- conv (WMMA GEMM) kernels ------------------------

__global__ __launch_bounds__(256) void k_atoms2h(const float* __restrict__ a,
                                                 _Float16* __restrict__ h) {
  int i = blockIdx.x * 256 + threadIdx.x;
  if (i < N_ATOMS * KERN) h[i] = (_Float16)a[i];
}

__global__ __launch_bounds__(256) void k_pad_half(const float* __restrict__ band,
                                                  _Float16* __restrict__ xpad,
                                                  int L, int total) {
  int i = blockIdx.x * 256 + threadIdx.x;
  if (i >= total) return;
  float v = (i >= 128 && i < 128 + L) ? band[i - 128] : 0.f;
  xpad[i] = (_Float16)v;
}

// fullT[t, a] = sum_k atoms[a,k] * xpad[t+k], a in [0,512), t in [0, L].
// Block: 8 waves = 128 rows; each wave sweeps NT=8 column tiles with A
// fragments held in registers. Grid: (ceil((L+1)/128), 4).
__global__ __launch_bounds__(256) void k_gemm_wmma(
    const _Float16* __restrict__ atomsH,   // 512 x 256
    const _Float16* __restrict__ xpad,     // >= L + 256 + pad, zero padded
    float* __restrict__ fullT,             // (L+1) x 512
    int L) {
  const int ncols = L + 1;
  const int lane = threadIdx.x & 31;
  const int wave = threadIdx.x >> 5;
  const int t0 = blockIdx.x * (16 * NT);
  const int aBase = blockIdx.y * 128 + wave * 16;
  const int m = lane & 15;   // A row-in-tile / B col-in-tile
  const int g = lane >> 4;   // lane group

  __shared__ _Float16 sx[SXW];

#if USE_ASYNC_LDS
  {
    const int nvec = SXW * (int)sizeof(_Float16) / 16;   // 48 x b128
    if (threadIdx.x < nvec) {
      v4i* gsrc = (v4i*)(xpad + t0) + threadIdx.x;   // C-cast drops const
      v4i* ldst = (v4i*)sx + threadIdx.x;
      __builtin_amdgcn_global_load_async_to_lds_b128(
          (__attribute__((address_space(1))) v4i*)gsrc,
          (__attribute__((address_space(3))) v4i*)ldst, 0, 0);
    }
    __builtin_amdgcn_s_wait_asynccnt(0);
    __syncthreads();
  }
#else
  for (int i = threadIdx.x; i < SXW; i += 256) sx[i] = xpad[t0 + i];
  __syncthreads();
#endif

  // A fragments, register resident for the whole column sweep.
  const _Float16* arow = atomsH + (size_t)(aBase + m) * KERN;
  v16h A[8];
#pragma unroll
  for (int s8 = 0; s8 < 8; ++s8) {
    const int kb = s8 * 32;
#pragma unroll
    for (int v = 0; v < 8; ++v) {
      int k0 = kb + ((v < 4) ? (2 * v + 8 * g) : (16 + 2 * (v - 4) + 8 * g));
      A[s8][2 * v]     = arow[k0];
      A[s8][2 * v + 1] = arow[k0 + 1];
    }
  }

#pragma unroll
  for (int tt = 0; tt < NT; ++tt) {
    v8f c = {};
#pragma unroll
    for (int s8 = 0; s8 < 8; ++s8) {
      v16h bf;
#pragma unroll
      for (int j = 0; j < 16; ++j)
        bf[j] = sx[tt * 16 + m + s8 * 32 + g * 8 + j + ((j >> 3) << 3)];
      c = __builtin_amdgcn_wmma_f32_16x16x32_f16(false, A[s8], false, bf,
                                                 (short)0, c, false, false);
    }
    const int col = t0 + tt * 16 + m;
    if (col < ncols) {
      float* dst = fullT + (size_t)col * N_ATOMS + aBase + 8 * g;  // 32B aligned
#pragma unroll
      for (int r = 0; r < 8; ++r) dst[r] = c[r];
    }
  }
}

// Two-stage deterministic column-mean of fullT: feat[a] = mean_t fullT[t,a].
__global__ __launch_bounds__(256) void k_mean1(const float* __restrict__ fullT,
                                               float* __restrict__ partial,
                                               int ncols) {
  int a = blockIdx.x * 256 + threadIdx.x;
  int chunk = blockIdx.y;
  int tpc = (ncols + 63) >> 6;
  int tbeg = chunk * tpc;
  int tend = tbeg + tpc; if (tend > ncols) tend = ncols;
  float acc = 0.f;
  for (int t = tbeg; t < tend; ++t) acc += fullT[(size_t)t * N_ATOMS + a];
  partial[chunk * N_ATOMS + a] = acc;
}

__global__ __launch_bounds__(256) void k_mean2(const float* __restrict__ partial,
                                               float* __restrict__ feat, int ncols) {
  int a = blockIdx.x * 256 + threadIdx.x;
  float acc = 0.f;
  for (int c = 0; c < 64; ++c) acc += partial[c * N_ATOMS + a];
  feat[a] = acc / (float)ncols;
}

// Destructive exact abs-top-64 (single block, 64 argmax passes).
__global__ __launch_bounds__(1024) void k_topk(float* __restrict__ fullT, int n,
                                               int* __restrict__ cidx,
                                               float* __restrict__ cval) {
  __shared__ float sv[1024];
  __shared__ int   si[1024];
  for (int it = 0; it < K_KEEP; ++it) {
    float best = -1.f; int bi = 0;
    for (int i = threadIdx.x; i < n; i += 1024) {
      float v = fabsf(fullT[i]);
      if (v > best) { best = v; bi = i; }
    }
    sv[threadIdx.x] = best; si[threadIdx.x] = bi; __syncthreads();
    for (int o = 512; o > 0; o >>= 1) {
      if (threadIdx.x < o) {
        float vo = sv[threadIdx.x + o];
        if (vo > sv[threadIdx.x] ||
            (vo == sv[threadIdx.x] && si[threadIdx.x + o] < si[threadIdx.x])) {
          sv[threadIdx.x] = vo; si[threadIdx.x] = si[threadIdx.x + o];
        }
      }
      __syncthreads();
    }
    if (threadIdx.x == 0) {
      int id = si[0];
      cidx[it] = id; cval[it] = fullT[id];
      fullT[id] = 0.f;
    }
    __syncthreads();
  }
}

// Transpose conv of the 64 surviving coefficients (fullT flat id = t*512 + a).
__global__ __launch_bounds__(256) void k_rec(float* __restrict__ rec,
                                             const int* __restrict__ cidx,
                                             const float* __restrict__ cval,
                                             const float* __restrict__ atoms,
                                             int L) {
  int t = blockIdx.x * 256 + threadIdx.x;
  if (t >= L) return;
  float acc = 0.f;
  for (int c = 0; c < K_KEEP; ++c) {
    int id = cidx[c];
    int a = id & (N_ATOMS - 1);
    int tau = id >> 9;
    int j = t + 128 - tau;
    if (j >= 0 && j < KERN) acc += cval[c] * atoms[a * KERN + j];
  }
  rec[t] = acc;
}

// ------------------------------- MLP heads ---------------------------------

struct MlpArgs {
  const float* W[8];   // 0..3 rooms head, 4..7 mix head
  const float* Bi[8];
  const float* feat;   // 3072
  float* rooms;        // 8 (softmaxed)
  float* mix;          // 1 (sigmoid)
};

__global__ __launch_bounds__(128) void k_mlp(MlpArgs A) {
  __shared__ float h0[FEAT_IN];
  __shared__ float h1[HID], h2[HID];
  const int tid = threadIdx.x;
  for (int i = tid; i < FEAT_IN; i += 128) h0[i] = A.feat[i];
  __syncthreads();
  for (int head = 0; head < 2; ++head) {
    const float* const* W = &A.W[head * 4];
    const float* const* Bb = &A.Bi[head * 4];
    { float acc = Bb[0][tid];
      for (int i = 0; i < FEAT_IN; ++i) acc += h0[i] * W[0][i * HID + tid];
      h1[tid] = acc > 0.f ? acc : 0.2f * acc; }
    __syncthreads();
    { float acc = Bb[1][tid];
      for (int i = 0; i < HID; ++i) acc += h1[i] * W[1][i * HID + tid];
      h2[tid] = acc > 0.f ? acc : 0.2f * acc; }
    __syncthreads();
    { float acc = Bb[2][tid];
      for (int i = 0; i < HID; ++i) acc += h2[i] * W[2][i * HID + tid];
      h1[tid] = acc > 0.f ? acc : 0.2f * acc; }
    __syncthreads();
    if (head == 0) {
      if (tid < 8) {
        float acc = Bb[3][tid];
        for (int i = 0; i < HID; ++i) acc += h1[i] * W[3][i * 8 + tid];
        h2[tid] = acc;
      }
      __syncthreads();
      if (tid == 0) {
        float mx = -1e30f;
        for (int r = 0; r < 8; ++r) mx = fmaxf(mx, h2[r]);
        float e[8]; float sum = 0.f;
        for (int r = 0; r < 8; ++r) { e[r] = __expf(h2[r] - mx); sum += e[r]; }
        for (int r = 0; r < 8; ++r) A.rooms[r] = e[r] / sum;
      }
      __syncthreads();
    } else {
      if (tid == 0) {
        float acc = Bb[3][0];
        for (int i = 0; i < HID; ++i) acc += h1[i] * W[3][i];
        A.mix[0] = 1.f / (1.f + __expf(-acc));
      }
    }
  }
}

__global__ __launch_bounds__(256) void k_rir(const float* __restrict__ rooms,
                                             const float* __restrict__ rooms_ir,
                                             float* __restrict__ rir, int n) {
  int i = blockIdx.x * 256 + threadIdx.x;
  int b = blockIdx.y;
  if (i >= n) return;
  float acc = 0.f;
  for (int r = 0; r < 8; ++r)
    acc += rooms[b * 8 + r] * rooms_ir[(size_t)r * n + i];
  rir[(size_t)b * n + i] = acc;
}

// --------------------------------- driver ----------------------------------

extern "C" void kernel_launch(void* const* d_in, const int* in_sizes, int n_in,
                              void* d_out, int out_size, void* d_ws, size_t ws_size,
                              hipStream_t stream) {
  (void)in_sizes; (void)n_in; (void)out_size; (void)ws_size;
  const int N = N_SAMP;
  const int BANDS[6] = {1024, 2048, 4096, 8192, 16384, 32768};

  const float* x = (const float*)d_in[0];
  const float* atoms[6];
  for (int i = 0; i < 6; ++i) atoms[i] = (const float*)d_in[1 + i];
  const float* rooms_ir = (const float*)d_in[7];

  // ---- workspace bump allocator ----
  size_t off = 0;
  auto alloc = [&](size_t bytes) -> char* {
    char* p = (char*)d_ws + off;
    off = (off + bytes + 255) & ~(size_t)255;
    return p;
  };
  cplx* cbufA = (cplx*)alloc((size_t)2 * N * sizeof(cplx));
  cplx* cbufB = (cplx*)alloc((size_t)2 * N * sizeof(cplx));
  cplx* spec  = (cplx*)alloc((size_t)(N / 2 + 1) * sizeof(cplx));
  cplx* acc   = (cplx*)alloc((size_t)(N / 2 + 1) * sizeof(cplx));
  int boff[7]; boff[0] = 0;
  for (int i = 0; i < 6; ++i) boff[i + 1] = boff[i] + BANDS[i];
  float* bands = (float*)alloc((size_t)BATCH * boff[6] * sizeof(float));
  float* recs  = (float*)alloc((size_t)BATCH * boff[6] * sizeof(float));
  _Float16* xpadH  = (_Float16*)alloc((size_t)(N + 512) * sizeof(_Float16));
  _Float16* atomsH = (_Float16*)alloc((size_t)N_ATOMS * KERN * sizeof(_Float16));
  float* fullT = (float*)alloc((size_t)(N + 1) * N_ATOMS * sizeof(float));
  float* partial = (float*)alloc((size_t)64 * N_ATOMS * sizeof(float));
  float* feats = (float*)alloc((size_t)BATCH * FEAT_IN * sizeof(float));
  float* rooms = (float*)alloc((size_t)BATCH * 8 * sizeof(float));
  float* mixv  = (float*)alloc((size_t)BATCH * sizeof(float));
  float* rir   = (float*)alloc((size_t)BATCH * N * sizeof(float));
  float* sig   = (float*)alloc((size_t)BATCH * N * sizeof(float));
  int*   cidx  = (int*)alloc(K_KEEP * sizeof(int));
  float* cval  = (float*)alloc(K_KEEP * sizeof(float));

  auto bandPtr = [&](int i, int b) { return bands + (size_t)BATCH * boff[i] + (size_t)b * BANDS[i]; };
  auto recPtr  = [&](int i, int b) { return recs  + (size_t)BATCH * boff[i] + (size_t)b * BANDS[i]; };

  auto fft = [&](cplx* buf, int n, float sign) {
    int lg = 0; while ((1 << lg) < n) ++lg;
    k_bitrev<<<dim3((n + 255) / 256), dim3(256), 0, stream>>>(buf, n, lg);
    for (int len = 2; len <= n; len <<= 1)
      k_fft_stage<<<dim3((n / 2 + 255) / 256), dim3(256), 0, stream>>>(buf, n, len, sign);
  };

  const float isqN = 1.f / sqrtf((float)N);

  // ---- phase 1: FFT band decomposition ----
  for (int b = 0; b < BATCH; ++b) {
    k_r2c<<<dim3((N + 255) / 256), dim3(256), 0, stream>>>(x + (size_t)b * N, cbufA, N, N);
    fft(cbufA, N, -1.f);
    (void)hipMemcpyAsync(spec, cbufA, (size_t)(N / 2 + 1) * sizeof(cplx),
                         hipMemcpyDeviceToDevice, stream);
    for (int i = 0; i < 6; ++i) {
      int s = BANDS[i];
      int mn = (i == 0) ? 0 : s / 4, mx = s / 2;
      k_band_spec<<<dim3((s + 255) / 256), dim3(256), 0, stream>>>(spec, cbufA, s, mn, mx, isqN);
      fft(cbufA, s, +1.f);
      k_c2r_scale<<<dim3((s + 255) / 256), dim3(256), 0, stream>>>(
          bandPtr(i, b), cbufA, s, 1.f / sqrtf((float)s));
    }
  }

  // ---- phase 2: per-band dictionary conv (WMMA) + features + top-k + rec ----
  for (int i = 0; i < 6; ++i) {
    const int s = BANDS[i], ncols = s + 1;
    k_atoms2h<<<dim3((N_ATOMS * KERN + 255) / 256), dim3(256), 0, stream>>>(atoms[i], atomsH);
    for (int b = 0; b < BATCH; ++b) {
      const int tot = s + 512;
      k_pad_half<<<dim3((tot + 255) / 256), dim3(256), 0, stream>>>(bandPtr(i, b), xpadH, s, tot);
      k_gemm_wmma<<<dim3((ncols + 16 * NT - 1) / (16 * NT), N_ATOMS / 128), dim3(256), 0, stream>>>(
          atomsH, xpadH, fullT, s);
      k_mean1<<<dim3(N_ATOMS / 256, 64), dim3(256), 0, stream>>>(fullT, partial, ncols);
      k_mean2<<<dim3(N_ATOMS / 256), dim3(256), 0, stream>>>(
          partial, feats + (size_t)b * FEAT_IN + i * N_ATOMS, ncols);
      k_topk<<<dim3(1), dim3(1024), 0, stream>>>(fullT, N_ATOMS * ncols, cidx, cval);
      k_rec<<<dim3((s + 255) / 256), dim3(256), 0, stream>>>(recPtr(i, b), cidx, cval, atoms[i], s);
    }
  }

  // ---- phase 3: MLP heads + room IR mix ----
  for (int b = 0; b < BATCH; ++b) {
    MlpArgs A;
    for (int l = 0; l < 4; ++l) {
      A.W[l]      = (const float*)d_in[8 + 2 * l];
      A.Bi[l]     = (const float*)d_in[9 + 2 * l];
      A.W[4 + l]  = (const float*)d_in[16 + 2 * l];
      A.Bi[4 + l] = (const float*)d_in[17 + 2 * l];
    }
    A.feat = feats + (size_t)b * FEAT_IN;
    A.rooms = rooms + (size_t)b * 8;
    A.mix = mixv + b;
    k_mlp<<<dim3(1), dim3(128), 0, stream>>>(A);
  }
  k_rir<<<dim3((N + 255) / 256, BATCH), dim3(256), 0, stream>>>(rooms, rooms_ir, rir, N);

  // ---- phase 4: recompose + room convolution + wet/dry mix ----
  const int M = 2 * N;
  for (int b = 0; b < BATCH; ++b) {
    k_zero_c<<<dim3((N / 2 + 1 + 255) / 256), dim3(256), 0, stream>>>(acc, N / 2 + 1);
    for (int i = 0; i < 5; ++i) {
      int s = BANDS[i];
      k_r2c<<<dim3((s + 255) / 256), dim3(256), 0, stream>>>(recPtr(i, b), cbufA, s, s);
      fft(cbufA, s, -1.f);
      k_spec_accum<<<dim3((s / 2 + 256) / 256), dim3(256), 0, stream>>>(
          cbufA, acc, s, 1.f / sqrtf((float)s));
    }
    k_herm<<<dim3((N + 255) / 256), dim3(256), 0, stream>>>(acc, cbufA, N);
    fft(cbufA, N, +1.f);
    k_signal<<<dim3((N + 255) / 256), dim3(256), 0, stream>>>(
        cbufA, recPtr(5, b), sig + (size_t)b * N, N, isqN);

    // FFT convolution with room IR (length 2N), ortho scaling folded in.
    k_r2c<<<dim3((M + 255) / 256), dim3(256), 0, stream>>>(sig + (size_t)b * N, cbufA, N, M);
    fft(cbufA, M, -1.f);
    k_r2c<<<dim3((M + 255) / 256), dim3(256), 0, stream>>>(rir + (size_t)b * N, cbufB, N, M);
    fft(cbufB, M, -1.f);
    k_cmul<<<dim3((M + 255) / 256), dim3(256), 0, stream>>>(cbufA, cbufB, M, 1.f / (float)M);
    fft(cbufA, M, +1.f);
    k_out<<<dim3((N + 255) / 256), dim3(256), 0, stream>>>(
        sig + (size_t)b * N, cbufA, mixv + b,
        (float*)d_out + (size_t)b * N, N, 1.f / sqrtf((float)M));
  }
}
